// Model_54820962566610
// MI455X (gfx1250) — matrix operand order repaired and checked by reading, compiled, not run on hardware
//
#include <hip/hip_runtime.h>
#include <hip/hip_bf16.h>

// ---------------------------------------------------------------------------
// Set Transformer forward for MI455X (gfx1250, wave32, WMMA).
//
// Roofline: ~0.6 TFLOP of GEMM math vs enormous bf16-WMMA ceiling -> the run
// is HBM-bound on the (64,2048,512) activations. Strategy:
//   * all activations stored bf16 (halves HBM traffic)
//   * all weights pre-converted once to bf16, transposed to (N,K) so GEMM
//     tile staging is raw byte movement
//   * GEMM operand tiles staged with GLOBAL_LOAD_ASYNC_TO_LDS_B128 (CDNA5
//     async global->LDS DMA, ASYNCcnt) -- no VGPR round trip, no conversion
//   * all GEMMs via v_wmma_f32_16x16x32_bf16, f32 accumulate
//   * flash attention (online softmax) -> scores never hit HBM
//   * bias / relu-residual fused into GEMM epilogues
// ---------------------------------------------------------------------------

#define BATCH   64
#define SETN    2048
#define DIN     64
#define DH      512
#define NHEADS  8
#define HD      64
#define NINDS   64
#define NSEEDS  4
#define DOUT    128
#define ATTN_SCALE 0.04419417382f   // 1/sqrt(512)

typedef float  v8f     __attribute__((ext_vector_type(8)));
typedef float  f32x4   __attribute__((ext_vector_type(4)));
typedef __bf16 bf16x16 __attribute__((ext_vector_type(16)));
typedef __bf16 bf16x8  __attribute__((ext_vector_type(8)));
typedef __bf16 bf16x4  __attribute__((ext_vector_type(4)));

__device__ __forceinline__ __bf16 f2bf(float f) { return (__bf16)f; }

__device__ __forceinline__ bf16x16 join8(bf16x8 lo, bf16x8 hi) {
    bf16x16 r;
#pragma unroll
    for (int i = 0; i < 8; ++i) { r[i] = lo[i]; r[i + 8] = hi[i]; }
    return r;
}

__device__ __forceinline__ v8f wmma_bf16(bf16x16 a, bf16x16 b, v8f c) {
    return __builtin_amdgcn_wmma_f32_16x16x32_bf16(
        false, a, false, b, (short)0, c, false, false);
}

// CDNA5 async global->LDS copy, 16B per lane (VGLOBAL GV mode, off saddr).
// VDST carries the per-lane LDS byte address, VADDR the 64-bit global addr.
__device__ __forceinline__ void async_ld_b128(void* lds_ptr, const void* gptr) {
    unsigned lds_off = (unsigned)(uintptr_t)lds_ptr;
    unsigned long long ga = (unsigned long long)(uintptr_t)gptr;
    asm volatile("global_load_async_to_lds_b128 %0, %1, off"
                 :: "v"(lds_off), "v"(ga) : "memory");
}
__device__ __forceinline__ void wait_async0() {
    asm volatile("s_wait_asynccnt 0" ::: "memory");
}

// ---------------------------------------------------------------------------
// Weight prep: f32 (K,N) row-major -> bf16 (N,K) row-major (one-time/launch)
// ---------------------------------------------------------------------------
__global__ __launch_bounds__(256) void wt_kernel(
    const float* __restrict__ W, __bf16* __restrict__ Wt, int K, int N)
{
    int idx = blockIdx.x * 256 + threadIdx.x;
    if (idx < K * N) {
        int k = idx / N, n = idx % N;
        Wt[(size_t)n * K + k] = f2bf(W[idx]);
    }
}

// ---------------------------------------------------------------------------
// Batched GEMM: C[b,m,n] = act(A[b,m,:] @ W[:,n] + bias[n])
//   mode 0: C = A@W + bias          mode 1: C = R + relu(A@W + bias)
// A: (B,M,K) bf16 or f32 (ABF), per-batch elem stride (0 = broadcast).
// Wt: bf16 (N,K) row-major (pre-transposed).  C: bf16 or f32 (CF32).
// Block 128 thr (4 waves), tile 64x64, BK=32. N % 64 == 0, K % 32 == 0.
// Full bf16 tiles are staged via async global->LDS DMA.
// ---------------------------------------------------------------------------
template<bool ABF, bool CF32>
__global__ __launch_bounds__(128) void gemm_kernel(
    const void* __restrict__ Ap, long long a_bstride,
    const __bf16* __restrict__ Wt, const float* __restrict__ bias,
    const __bf16* __restrict__ R, int mode,
    void* __restrict__ Cp, int M, int K, int N)
{
    const int b     = blockIdx.z;
    const int tileM = blockIdx.y * 64;
    const int tileN = blockIdx.x * 64;
    const int tid   = threadIdx.x;
    const int lane  = tid & 31;
    const int wid   = tid >> 5;
    const bool afull = (tileM + 64 <= M);

    __shared__ __bf16 As[64][32];   // [m][k]
    __shared__ __bf16 Ws[64][32];   // [n][k]

    v8f acc[4];
#pragma unroll
    for (int t = 0; t < 4; ++t) acc[t] = {};

    for (int k0 = 0; k0 < K; k0 += 32) {
        // ---- stage A tile ----
        if (ABF && afull) {
            const __bf16* Ab = (const __bf16*)Ap + (size_t)b * (size_t)a_bstride;
#pragma unroll
            for (int r = 0; r < 2; ++r) {
                int idx = tid + r * 128;
                int row = idx >> 2, col = (idx & 3) * 8;
                async_ld_b128(&As[row][col],
                              Ab + (size_t)(tileM + row) * K + k0 + col);
            }
        } else if (ABF) {
            const __bf16* Ab = (const __bf16*)Ap + (size_t)b * (size_t)a_bstride;
#pragma unroll
            for (int r = 0; r < 2; ++r) {
                int idx = tid + r * 128;
                int row = idx >> 2, col = (idx & 3) * 8;
                int grow = tileM + row;
                int gr   = grow < M ? grow : M - 1;
                bf16x8 v = *(const bf16x8*)(Ab + (size_t)gr * K + k0 + col);
                bf16x8 z = {};
                if (grow >= M) v = z;
                *(bf16x8*)&As[row][col] = v;
            }
        } else {
            const float* Ab = (const float*)Ap + (size_t)b * (size_t)a_bstride;
#pragma unroll
            for (int r = 0; r < 4; ++r) {
                int idx = tid + r * 128;
                int row = idx >> 3, col = (idx & 7) * 4;
                int grow = tileM + row;
                int gr   = grow < M ? grow : M - 1;
                f32x4 v = *(const f32x4*)(Ab + (size_t)gr * K + k0 + col);
                bf16x4 o;
#pragma unroll
                for (int j = 0; j < 4; ++j)
                    o[j] = f2bf(grow < M ? v[j] : 0.f);
                *(bf16x4*)&As[row][col] = o;
            }
        }
        // ---- stage W tile (always full, always async) ----
#pragma unroll
        for (int r = 0; r < 2; ++r) {
            int idx = tid + r * 128;
            int n = idx >> 2, kc = (idx & 3) * 8;
            async_ld_b128(&Ws[n][kc],
                          Wt + (size_t)(tileN + n) * K + k0 + kc);
        }
        wait_async0();
        __syncthreads();

        // A fragment: lane = row, K = {kb..kb+7, kb+16..kb+23}, kb=(lane/16)*8
        int arow = wid * 16 + (lane & 15);
        int kb   = (lane >> 4) * 8;
        bf16x16 afrag = join8(*(const bf16x8*)&As[arow][kb],
                              *(const bf16x8*)&As[arow][kb + 16]);
#pragma unroll
        for (int t = 0; t < 4; ++t) {
            int bcol = t * 16 + (lane & 15);
            int kb2  = (lane >> 4) * 8;
            bf16x16 bfrag = join8(*(const bf16x8*)&Ws[bcol][kb2],
                                  *(const bf16x8*)&Ws[bcol][kb2 + 16]);
            acc[t] = wmma_bf16(afrag, bfrag, acc[t]);
        }
        __syncthreads();
    }

    // ---- epilogue: bias (+ relu-residual), C layout stores ----
    const int rbase = tileM + wid * 16 + ((lane >> 4) << 3);
    const int cbase = tileN + (lane & 15);
#pragma unroll
    for (int t = 0; t < 4; ++t) {
#pragma unroll
        for (int i = 0; i < 8; ++i) {
            int row = rbase + i;
            int col = cbase + t * 16;
            if (row < M) {
                float v = acc[t][i] + bias[col];
                size_t oidx = ((size_t)b * M + row) * N + col;
                if (mode == 1) v = (float)R[oidx] + fmaxf(v, 0.f);
                if (CF32) ((float*)Cp)[oidx] = v;
                else      ((__bf16*)Cp)[oidx] = f2bf(v);
            }
        }
    }
}

// ---------------------------------------------------------------------------
// Flash attention, one wave32 per (batch, head, 16-row Q tile).
// O[b,q,:] = Qp[b,q,:] + softmax(Qh Kh^T * scale) @ Vh   (head dim 64)
// ---------------------------------------------------------------------------
__global__ __launch_bounds__(32) void attn_kernel(
    const __bf16* __restrict__ Qp, const __bf16* __restrict__ Kp,
    const __bf16* __restrict__ Vp, __bf16* __restrict__ O,
    int nq, int nk)
{
    const int b    = blockIdx.z;
    const int hd0  = blockIdx.y * HD;
    const int q0   = blockIdx.x * 16;
    const int lane = threadIdx.x;
    const int lr   = lane & 15;
    const int lh   = lane >> 4;

    __shared__ __bf16 Ps[16][32];
    const bf16x8 zero8 = {};

    bf16x16 qf[2];
    {
        int  qrow = q0 + lr;
        int  qc   = qrow < nq ? qrow : nq - 1;
        const __bf16* qptr = Qp + ((size_t)b * nq + qc) * DH + hd0;
        int kb = lh * 8;
#pragma unroll
        for (int s = 0; s < 2; ++s) {
            bf16x8 lo = *(const bf16x8*)(qptr + s * 32 + kb);
            bf16x8 hi = *(const bf16x8*)(qptr + s * 32 + kb + 16);
            if (qrow >= nq) { lo = zero8; hi = zero8; }
            qf[s] = join8(lo, hi);
        }
    }

    v8f oacc[4];
#pragma unroll
    for (int t = 0; t < 4; ++t) oacc[t] = {};
    float mrow[8], lrow[8];
#pragma unroll
    for (int i = 0; i < 8; ++i) { mrow[i] = -1e30f; lrow[i] = 0.f; }

    for (int kt = 0; kt < nk; kt += 32) {
        v8f sfr[2];
#pragma unroll
        for (int t = 0; t < 2; ++t) {
            v8f s_acc = {};
            int key = kt + t * 16 + lr;
            int kc  = key < nk ? key : nk - 1;
#pragma unroll
            for (int s = 0; s < 2; ++s) {
                const __bf16* kp = Kp + ((size_t)b * nk + kc) * DH
                                   + hd0 + s * 32 + lh * 8;
                bf16x8 lo = *(const bf16x8*)kp;
                bf16x8 hi = *(const bf16x8*)(kp + 16);
                if (key >= nk) { lo = zero8; hi = zero8; }
                s_acc = wmma_bf16(qf[s], join8(lo, hi), s_acc);
            }
            sfr[t] = s_acc;
        }

#pragma unroll
        for (int i = 0; i < 8; ++i) {
            float a0 = sfr[0][i] * ATTN_SCALE;
            float a1 = sfr[1][i] * ATTN_SCALE;
            if (kt + lr >= nk)      a0 = -1e30f;
            if (kt + 16 + lr >= nk) a1 = -1e30f;
            float mx = fmaxf(a0, a1);
#pragma unroll
            for (int off = 8; off >= 1; off >>= 1)
                mx = fmaxf(mx, __shfl_xor(mx, off, 32));
            float mnew = fmaxf(mrow[i], mx);
            float corr = __expf(mrow[i] - mnew);
            float p0 = __expf(a0 - mnew);
            float p1 = __expf(a1 - mnew);
            float ps = p0 + p1;
#pragma unroll
            for (int off = 8; off >= 1; off >>= 1)
                ps += __shfl_xor(ps, off, 32);
            lrow[i] = lrow[i] * corr + ps;
            mrow[i] = mnew;
#pragma unroll
            for (int t = 0; t < 4; ++t) oacc[t][i] *= corr;
            int pr = i + lh * 8;
            Ps[pr][lr]      = f2bf(p0);
            Ps[pr][16 + lr] = f2bf(p1);
        }
        asm volatile("s_wait_dscnt 0" ::: "memory");  // LDS store->load, 1 wave

        bf16x16 pfrag = join8(*(const bf16x8*)&Ps[lr][lh * 8],
                              *(const bf16x8*)&Ps[lr][lh * 8 + 16]);

#pragma unroll
        for (int t = 0; t < 4; ++t) {
            bf16x16 vf;
            const __bf16* vbase = Vp + (size_t)b * nk * DH + hd0 + t * 16 + lr;
#pragma unroll
            for (int j = 0; j < 16; ++j) {
                int key = kt + lh * 8 + j + (j >= 8 ? 8 : 0);
                int kc  = key < nk ? key : nk - 1;
                __bf16 val = vbase[(size_t)kc * DH];
                vf[j] = key < nk ? val : (__bf16)0.f;
            }
            oacc[t] = wmma_bf16(pfrag, vf, oacc[t]);
        }
    }

#pragma unroll
    for (int t = 0; t < 4; ++t)
#pragma unroll
        for (int i = 0; i < 8; ++i) {
            int row = q0 + i + lh * 8;
            if (row < nq) {
                int col = hd0 + t * 16 + lr;
                size_t idx = ((size_t)b * nq + row) * DH + col;
                O[idx] = f2bf((float)Qp[idx] + oacc[t][i] / lrow[i]);
            }
        }
}

// ---------------------------------------------------------------------------
// Host-side orchestration
// ---------------------------------------------------------------------------
struct MabW  { const float *qw,*qb,*kw,*kb,*vw,*vb,*ow,*ob; };
struct MabWT { const __bf16 *qw,*kw,*vw,*ow; const float *qb,*kb,*vb,*ob; };

static inline MabW mabw(void* const* d, int base, bool sorted_order) {
    MabW m;
    if (sorted_order) {  // jax tree_flatten: k.b,k.w,o.b,o.w,q.b,q.w,v.b,v.w
        m.kb=(const float*)d[base+0]; m.kw=(const float*)d[base+1];
        m.ob=(const float*)d[base+2]; m.ow=(const float*)d[base+3];
        m.qb=(const float*)d[base+4]; m.qw=(const float*)d[base+5];
        m.vb=(const float*)d[base+6]; m.vw=(const float*)d[base+7];
    } else {             // insertion order: q.w,q.b,k.w,k.b,v.w,v.b,o.w,o.b
        m.qw=(const float*)d[base+0]; m.qb=(const float*)d[base+1];
        m.kw=(const float*)d[base+2]; m.kb=(const float*)d[base+3];
        m.vw=(const float*)d[base+4]; m.vb=(const float*)d[base+5];
        m.ow=(const float*)d[base+6]; m.ob=(const float*)d[base+7];
    }
    return m;
}

static inline void launch_gemm(const void* A, bool a_bf, long long a_bstride,
                               const __bf16* Wt, const float* bias,
                               const __bf16* R, int mode,
                               void* C, bool c_f32,
                               int M, int K, int N, hipStream_t s)
{
    dim3 g(N / 64, (M + 63) / 64, BATCH);
    if (a_bf) {
        if (c_f32) gemm_kernel<true, true ><<<g, 128, 0, s>>>(A, a_bstride, Wt, bias, R, mode, C, M, K, N);
        else       gemm_kernel<true, false><<<g, 128, 0, s>>>(A, a_bstride, Wt, bias, R, mode, C, M, K, N);
    } else {
        gemm_kernel<false, false><<<g, 128, 0, s>>>(A, a_bstride, Wt, bias, R, mode, C, M, K, N);
    }
}

static inline void run_mab(const void* Qin, bool qbf, long long q_bs, int nq, int dq,
                           const void* Kin, bool kbf, long long k_bs, int nk, int dk,
                           const MabWT& w,
                           __bf16* Qp, __bf16* Kp, __bf16* Vp, __bf16* Obuf, __bf16* Out,
                           hipStream_t s)
{
    launch_gemm(Qin, qbf, q_bs, w.qw, w.qb, nullptr, 0, Qp, false, nq, dq, DH, s);
    launch_gemm(Kin, kbf, k_bs, w.kw, w.kb, nullptr, 0, Kp, false, nk, dk, DH, s);
    launch_gemm(Kin, kbf, k_bs, w.vw, w.vb, nullptr, 0, Vp, false, nk, dk, DH, s);
    attn_kernel<<<dim3((nq + 15) / 16, NHEADS, BATCH), 32, 0, s>>>(Qp, Kp, Vp, Obuf, nq, nk);
    launch_gemm(Obuf, true, (long long)nq * DH, w.ow, w.ob, Obuf, 1, Out, false, nq, DH, DH, s);
}

extern "C" void kernel_launch(void* const* d_in, const int* in_sizes, int n_in,
                              void* d_out, int out_size, void* d_ws, size_t ws_size,
                              hipStream_t stream)
{
    (void)n_in; (void)out_size; (void)ws_size;
    const float* X = (const float*)d_in[0];
    auto F = [&](int i){ return (const float*)d_in[i]; };

    // Disambiguate flatten order: sorted puts isab1.mab0.k.b (512) at index 2;
    // insertion order puts isab1.mab0.q.w (262144) there.
    const bool so = (in_sizes[2] == DH);

    const float *I1, *I2, *S, *outw, *outb;
    int b_pma, b_s1, b_s2;
    I1 = F(1); I2 = F(18);
    if (so) {
        outb = F(35); outw = F(36);
        S = F(37); b_pma = 38; b_s1 = 46; b_s2 = 54;
    } else {
        S = F(35); b_pma = 36; b_s1 = 44; b_s2 = 52;
        outw = F(60); outb = F(61);
    }
    MabW i1m0 = mabw(d_in, 2, so),  i1m1 = mabw(d_in, 10, so);
    MabW i2m0 = mabw(d_in, 19, so), i2m1 = mabw(d_in, 27, so);
    MabW pmam = mabw(d_in, b_pma, so);
    MabW s1w  = mabw(d_in, b_s1, so), s2w = mabw(d_in, b_s2, so);

    // ---- workspace layout: [bf16 weights | 3 big | 6 small] ----
    __bf16* cursor = (__bf16*)d_ws;
    auto conv = [&](const float* W, int K, int N) -> const __bf16* {
        __bf16* dst = cursor; cursor += (size_t)K * N;
        int total = K * N;
        wt_kernel<<<(total + 255) / 256, 256, 0, stream>>>(W, dst, K, N);
        return dst;
    };
    auto make_mt = [&](const MabW& m, int dq, int dk) -> MabWT {
        MabWT t;
        t.qw = conv(m.qw, dq, DH); t.kw = conv(m.kw, dk, DH);
        t.vw = conv(m.vw, dk, DH); t.ow = conv(m.ow, DH, DH);
        t.qb = m.qb; t.kb = m.kb; t.vb = m.vb; t.ob = m.ob;
        return t;
    };

    MabWT t_i1m0 = make_mt(i1m0, DH, DIN);
    MabWT t_i1m1 = make_mt(i1m1, DIN, DH);
    MabWT t_i2m0 = make_mt(i2m0, DH, DH);
    MabWT t_i2m1 = make_mt(i2m1, DH, DH);
    MabWT t_pma  = make_mt(pmam, DH, DH);
    MabWT t_s1   = make_mt(s1w,  DH, DH);
    MabWT t_s2   = make_mt(s2w,  DH, DH);
    const __bf16* outwt = conv(outw, DH, DOUT);

    // activation buffers (bf16): 3 big (B,2048,512) + 6 small (B,64,512)
    const size_t BIGF = (size_t)BATCH * SETN * DH;
    const size_t SMF  = (size_t)BATCH * NINDS * DH;
    // align cursor to 16 elements
    cursor = (__bf16*)(((uintptr_t)cursor + 31) & ~(uintptr_t)31);
    __bf16* big0 = cursor;
    __bf16* big1 = big0 + BIGF;
    __bf16* big2 = big1 + BIGF;
    __bf16* sm0  = big2 + BIGF;
    __bf16* sm1  = sm0 + SMF;
    __bf16* sm2  = sm1 + SMF;
    __bf16* sm3  = sm2 + SMF;
    __bf16* sm4  = sm3 + SMF;
    __bf16* sm5  = sm4 + SMF;

    const long long XS = (long long)SETN * DIN;   // X batch stride (f32 input)
    const long long HS = (long long)NINDS * DH;
    const long long BS = (long long)SETN * DH;
    const long long TS = (long long)NSEEDS * DH;

    // ----- isab1 -----
    run_mab(I1, false, 0, NINDS, DH,   X, false, XS, SETN, DIN, t_i1m0,
            sm0, big0, big1, sm1, sm2, stream);                       // H1 = MAB(I1, X)
    run_mab(X, false, XS, SETN, DIN,   sm2, true, HS, NINDS, DH, t_i1m1,
            big0, sm0, sm1, big1, big2, stream);                      // h1 = MAB(X, H1)

    // ----- isab2 (input big2) -----
    run_mab(I2, false, 0, NINDS, DH,   big2, true, BS, SETN, DH, t_i2m0,
            sm0, big0, big1, sm1, sm2, stream);
    run_mab(big2, true, BS, SETN, DH,  sm2, true, HS, NINDS, DH, t_i2m1,
            big0, sm3, sm4, big1, big2, stream);

    // ----- PMA (input big2) -----
    run_mab(S, false, 0, NSEEDS, DH,   big2, true, BS, SETN, DH, t_pma,
            sm0, big0, big1, sm1, sm2, stream);

    // ----- SAB1 / SAB2 -----
    run_mab(sm2, true, TS, NSEEDS, DH, sm2, true, TS, NSEEDS, DH, t_s1,
            sm0, sm3, sm4, sm1, sm5, stream);
    run_mab(sm5, true, TS, NSEEDS, DH, sm5, true, TS, NSEEDS, DH, t_s2,
            sm0, sm3, sm4, sm1, sm2, stream);

    // ----- output projection (bf16 A -> f32 out) -----
    launch_gemm(sm2, true, TS, outwt, outb, nullptr, 0,
                d_out, true, NSEEDS, DH, DOUT, stream);
}
